// EGNNRegressor_7138235646498
// MI455X (gfx1250) — matrix operand order repaired and checked by reading, not compiled, not run on hardware
//
#include <hip/hip_runtime.h>
#include <hip/hip_bf16.h>

typedef __attribute__((ext_vector_type(16))) _Float16 v16h;
typedef __attribute__((ext_vector_type(8)))  _Float16 v8h;
typedef __attribute__((ext_vector_type(8)))  float    v8f;

#define NTOT   200000
#define NGRAPH 1000
#define NPG    200
#define KNN    5
#define MD     128
#define RT     10            // row tiles (of 16) per GEMM block -> 160 rows/block
#define WPAD   136           // sWt row stride in halves (128 + 8 pad -> 272 B, 16B aligned)

// ---------------------------------------------------------------------------
// KNN: one block per graph. Positions cached in LDS, each thread owns a node
// and keeps a sorted top-5 (smallest d2) in registers.
// ---------------------------------------------------------------------------
__global__ void knn_kernel(const float* __restrict__ pos, int* __restrict__ nbr) {
    __shared__ float sx[NPG], sy[NPG], sz[NPG];
    const int g = blockIdx.x;
    const float* p = pos + (size_t)g * NPG * 3;
    for (int i = threadIdx.x; i < NPG; i += blockDim.x) {
        sx[i] = p[i * 3 + 0];
        sy[i] = p[i * 3 + 1];
        sz[i] = p[i * 3 + 2];
    }
    __syncthreads();
    const int t = threadIdx.x;
    if (t < NPG) {
        float bd[KNN];
        int   bi[KNN];
#pragma unroll
        for (int i = 0; i < KNN; ++i) { bd[i] = 3.0e38f; bi[i] = 0; }
        const float x = sx[t], y = sy[t], z = sz[t];
        for (int j = 0; j < NPG; ++j) {
            if (j == t) continue;
            const float dx = sx[j] - x, dy = sy[j] - y, dz = sz[j] - z;
            const float d = dx * dx + dy * dy + dz * dz;
            if (d < bd[KNN - 1]) {
                bd[KNN - 1] = d; bi[KNN - 1] = j;
#pragma unroll
                for (int i = KNN - 1; i > 0; --i) {
                    if (bd[i] < bd[i - 1]) {
                        float td = bd[i]; bd[i] = bd[i - 1]; bd[i - 1] = td;
                        int   ti = bi[i]; bi[i] = bi[i - 1]; bi[i - 1] = ti;
                    }
                }
            }
        }
        int* o = nbr + ((size_t)g * NPG + t) * KNN;
#pragma unroll
        for (int i = 0; i < KNN; ++i) o[i] = g * NPG + bi[i];
    }
}

// ---------------------------------------------------------------------------
// Embedding gather, 8 features per thread: x16[node, 8f..8f+7] = f16(emb[z[node], ...])
// ---------------------------------------------------------------------------
__global__ void embed_kernel(const int* __restrict__ z, const float* __restrict__ emb,
                             _Float16* __restrict__ x16) {
    const int gid  = blockIdx.x * blockDim.x + threadIdx.x;   // over NTOT*16
    const int node = gid >> 4;
    const int seg  = gid & 15;
    const float* src = emb + (size_t)z[node] * MD + seg * 8;
    const float4 a = ((const float4*)src)[0];
    const float4 b = ((const float4*)src)[1];
    union { uint4 q; v8h h; } o;
    o.h[0] = (_Float16)a.x; o.h[1] = (_Float16)a.y;
    o.h[2] = (_Float16)a.z; o.h[3] = (_Float16)a.w;
    o.h[4] = (_Float16)b.x; o.h[5] = (_Float16)b.y;
    o.h[6] = (_Float16)b.z; o.h[7] = (_Float16)b.w;
    *(uint4*)(x16 + (size_t)node * MD + seg * 8) = o.q;
}

// ---------------------------------------------------------------------------
// Convert + transpose conv weights: Wt[l][col][k] = f16(W[l][k][col]).
// Column-major layout makes each WMMA B-fragment 16 contiguous halves.
// ---------------------------------------------------------------------------
__global__ void wcvt_kernel(const float* __restrict__ w, _Float16* __restrict__ wt, int n) {
    const int gid = blockIdx.x * blockDim.x + threadIdx.x;
    if (gid < n) {
        const int l = gid >> 14;
        const int r = gid & 16383;
        const int k = r >> 7;
        const int j = r & 127;
        wt[(size_t)l * MD * MD + (size_t)j * MD + k] = (_Float16)w[gid];
    }
}

// ---------------------------------------------------------------------------
// GEMM H = X @ W (f16 in, f32 WMMA accumulate, f16 out).
// Block = 256 threads (8 waves); each block owns 160 rows (10 tiles of 16).
// W (column-major) staged once per block; B fragments hoisted to registers.
// A tile staged in *fragment order*: global 16B chunk (row m, 8 halves
// k=8j..8j+7) lands at sAf[kk][hi*16+m][8*top], so each lane's A fragment is
// one contiguous 32 B LDS read.  chunk j: kk = j/4, hi = (j%4)/2, top = j%2
// Double-buffered + software-pipelined: the global load for tile t+1 is
// issued before computing tile t; its LDS store targets the opposite buffer,
// so only ONE barrier per tile is needed and load latency hides under WMMA.
// C/D layout: VGPR i -> row (lane<16 ? i : i+8), col = wave*16 + lane%16.
// ---------------------------------------------------------------------------
__global__ void gemm_kernel(const _Float16* __restrict__ X,   // [NTOT, 128]
                            const _Float16* __restrict__ Wt,  // [128 cols][128 k] col-major
                            _Float16* __restrict__ H) {       // [NTOT, 128]
    __shared__ __align__(32) _Float16 sAf[2][4 * 32 * 16];    // 2 x 4 KB, fragment order
    __shared__ __align__(32) _Float16 sWt[MD * WPAD];         // 34 KB, col-major + pad

    const int tid  = threadIdx.x;
    const int wave = tid >> 5;
    const int lane = tid & 31;
    const int m    = lane & 15;
    const int hi   = lane >> 4;
    const int nc   = (wave << 4) + m;          // output column

    // Stage W column-major with 16 B row pad (2048 x 16 B chunks, 8/thread).
#pragma unroll
    for (int i = 0; i < 8; ++i) {
        const int c = tid + i * 256;
        const int col = c >> 4, wdw = c & 15;
        *(uint4*)(sWt + col * WPAD + wdw * 8) = ((const uint4*)Wt)[c];
    }

    // Per-thread fragment-order destination for A-tile chunks (tile-invariant).
    const int mrow = tid >> 4, j = tid & 15;
    const int dstOff = ((j >> 2) * 32 + ((j >> 1) & 1) * 16 + mrow) * 16 + (j & 1) * 8;

    const size_t rowBase = (size_t)blockIdx.x * (RT * 16);

    // Prologue: stage tile 0 into buffer 0.
    uint4 chunk = ((const uint4*)(X + rowBase * MD))[tid];
    *(uint4*)(&sAf[0][0] + dstOff) = chunk;
    __syncthreads();                           // covers sWt + sAf[0]

    // Hoist B fragments (constant across all row tiles of this block).
    v16h bfrag[4];
#pragma unroll
    for (int kk = 0; kk < 4; ++kk) {
        union { v16h v; uint4 q[2]; } u;
        const uint4* bp = (const uint4*)(sWt + nc * WPAD + kk * 32 + hi * 16);
        u.q[0] = bp[0];
        u.q[1] = bp[1];
        bfrag[kk] = u.v;
    }

    int buf = 0;
    for (int t = 0; t < RT; ++t) {
        const size_t row0 = rowBase + (size_t)t * 16;

        if (t + 1 < RT) {
            // Prefetch tile t+2 and issue the tile t+1 load now; its latency
            // hides behind this tile's WMMAs and stores.
            __builtin_prefetch(X + (row0 + 32) * MD + tid * 8, 0, 1);
            chunk = ((const uint4*)(X + (row0 + 16) * MD))[tid];
        }

        v8f acc = {};
#pragma unroll
        for (int kk = 0; kk < 4; ++kk) {
            const v16h a = *(const v16h*)(&sAf[buf][0] + (kk * 32 + lane) * 16);
            acc = __builtin_amdgcn_wmma_f32_16x16x32_f16(
                false, a, false, bfrag[kk], (short)0, acc, false, false);
        }

        _Float16* hp = H + row0 * MD + nc;
#pragma unroll
        for (int i = 0; i < 8; ++i) {
            const int M = hi ? (i + 8) : i;
            hp[(size_t)M * MD] = (_Float16)acc[i];
        }

        if (t + 1 < RT) {
            // Stage tile t+1 into the opposite buffer; readers of sAf[buf]
            // are done (same thread), writers never touch sAf[buf].
            *(uint4*)(&sAf[buf ^ 1][0] + dstOff) = chunk;
            __syncthreads();
            buf ^= 1;
        }
    }
}

// ---------------------------------------------------------------------------
// GCN aggregation, 8 features per thread. knn_graph gives every node exactly
// K=5 in-edges, so deg = 6 uniformly and the symmetric norm is a constant:
//   x' = relu( (h[self] + sum_{j<5} h[nbr_j]) / 6 + b )
// ---------------------------------------------------------------------------
__global__ void aggregate_kernel(const _Float16* __restrict__ h16,
                                 const int* __restrict__ nbr,
                                 const float* __restrict__ bias,
                                 _Float16* __restrict__ x16) {
    const int gid  = blockIdx.x * blockDim.x + threadIdx.x;   // over NTOT*16
    const int node = gid >> 4;
    const int seg  = gid & 15;
    const int fo   = seg * 8;

    union { uint4 q; v8h h; } v;
    v.q = *(const uint4*)(h16 + (size_t)node * MD + fo);
    float acc[8];
#pragma unroll
    for (int e = 0; e < 8; ++e) acc[e] = (float)v.h[e];

    const int* nb = nbr + (size_t)node * KNN;
#pragma unroll
    for (int j = 0; j < KNN; ++j) {
        v.q = *(const uint4*)(h16 + (size_t)nb[j] * MD + fo);
#pragma unroll
        for (int e = 0; e < 8; ++e) acc[e] += (float)v.h[e];
    }

    union { uint4 q; v8h h; } o;
#pragma unroll
    for (int e = 0; e < 8; ++e)
        o.h[e] = (_Float16)fmaxf(acc[e] * (1.0f / 6.0f) + bias[fo + e], 0.0f);
    *(uint4*)(x16 + (size_t)node * MD + fo) = o.q;
}

// ---------------------------------------------------------------------------
// Mean pooling per graph: pooled[g, f] = mean_n x16[g*200+n, f]
// ---------------------------------------------------------------------------
__global__ void pool_kernel(const _Float16* __restrict__ x16, float* __restrict__ pooled) {
    const int g = blockIdx.x;
    const int f = threadIdx.x;     // 128
    float s = 0.0f;
    const _Float16* base = x16 + (size_t)g * NPG * MD + f;
    for (int n = 0; n < NPG; ++n) s += (float)base[(size_t)n * MD];
    pooled[(size_t)g * MD + f] = s * (1.0f / (float)NPG);
}

// ---------------------------------------------------------------------------
// Regressor MLP 128 -> 64 -> 32 -> 1, one block (128 threads) per graph.
// ---------------------------------------------------------------------------
__global__ void mlp_kernel(const float* __restrict__ pooled,
                           const float* __restrict__ W1, const float* __restrict__ b1,
                           const float* __restrict__ W2, const float* __restrict__ b2,
                           const float* __restrict__ W3, const float* __restrict__ b3,
                           float* __restrict__ out) {
    __shared__ float sp[MD], h1[64], h2[32];
    const int g = blockIdx.x;
    const int t = threadIdx.x;     // 128
    sp[t] = pooled[(size_t)g * MD + t];
    __syncthreads();
    if (t < 64) {
        float s = b1[t];
        for (int k = 0; k < MD; ++k) s += sp[k] * W1[k * 64 + t];
        h1[t] = fmaxf(s, 0.0f);
    }
    __syncthreads();
    if (t < 32) {
        float s = b2[t];
        for (int k = 0; k < 64; ++k) s += h1[k] * W2[k * 32 + t];
        h2[t] = fmaxf(s, 0.0f);
    }
    __syncthreads();
    if (t == 0) {
        float s = b3[0];
        for (int k = 0; k < 32; ++k) s += h2[k] * W3[k];
        out[g] = s;
    }
}

// ---------------------------------------------------------------------------
// Launcher
// ---------------------------------------------------------------------------
extern "C" void kernel_launch(void* const* d_in, const int* in_sizes, int n_in,
                              void* d_out, int out_size, void* d_ws, size_t ws_size,
                              hipStream_t stream) {
    (void)in_sizes; (void)n_in; (void)out_size; (void)ws_size;

    const int*   z     = (const int*)  d_in[0];
    const float* pos   = (const float*)d_in[1];
    // d_in[2] = batch (implicit: node / NPG), unused
    const float* emb   = (const float*)d_in[3];
    const float* convW = (const float*)d_in[4];
    const float* convb = (const float*)d_in[5];
    const float* rW1   = (const float*)d_in[6];
    const float* rb1   = (const float*)d_in[7];
    const float* rW2   = (const float*)d_in[8];
    const float* rb2   = (const float*)d_in[9];
    const float* rW3   = (const float*)d_in[10];
    const float* rb3   = (const float*)d_in[11];
    float* out = (float*)d_out;

    // Workspace layout (256B-aligned slabs)
    char* ws = (char*)d_ws;
    size_t off = 0;
    auto alloc = [&](size_t bytes) -> void* {
        void* p = ws + off;
        off += (bytes + 255) & ~(size_t)255;
        return p;
    };
    _Float16* x16    = (_Float16*)alloc((size_t)NTOT * MD * sizeof(_Float16)); // 51.2 MB
    _Float16* h16    = (_Float16*)alloc((size_t)NTOT * MD * sizeof(_Float16)); // 51.2 MB
    int*      nbr    = (int*)     alloc((size_t)NTOT * KNN * sizeof(int));     //  4.0 MB
    _Float16* W16t   = (_Float16*)alloc((size_t)3 * MD * MD * sizeof(_Float16));
    float*    pooled = (float*)   alloc((size_t)NGRAPH * MD * sizeof(float));

    // 1) KNN graph construction
    knn_kernel<<<NGRAPH, 256, 0, stream>>>(pos, nbr);

    // 2) Embedding gather (fp32 -> f16), 8 features/thread
    embed_kernel<<<(NTOT * 16) / 256, 256, 0, stream>>>(z, emb, x16);

    // 3) Conv weights -> f16, transposed to column-major
    const int wn = 3 * MD * MD;
    wcvt_kernel<<<(wn + 255) / 256, 256, 0, stream>>>(convW, W16t, wn);

    // 4) Three GCN layers: WMMA GEMM (pipelined, W reused over 10 tiles) + aggregation
    for (int l = 0; l < 3; ++l) {
        gemm_kernel<<<NTOT / (RT * 16), 256, 0, stream>>>(
            x16, W16t + (size_t)l * MD * MD, h16);
        aggregate_kernel<<<(NTOT * 16) / 256, 256, 0, stream>>>(
            h16, nbr, convb + (size_t)l * MD, x16);
    }

    // 5) Mean pooling per graph
    pool_kernel<<<NGRAPH, MD, 0, stream>>>(x16, pooled);

    // 6) Regressor head
    mlp_kernel<<<NGRAPH, MD, 0, stream>>>(pooled, rW1, rb1, rW2, rb2, rW3, rb3, out);
}
